// AdditiveAttention_2095944040803
// MI455X (gfx1250) — compile-verified, matrix operand order checked by reading, not run on hardware
//
#include <hip/hip_runtime.h>
#include <hip/hip_bf16.h>

typedef __attribute__((ext_vector_type(16))) _Float16 v16h;
typedef __attribute__((ext_vector_type(8)))  float    v8f;

#define NEG_INF_F (-1e6f)

// K-index inside a 32-wide K-slice for 16-bit WMMA A/B operands (wave32).
// From CDNA5 ISA 7.12.2: lanes 0-15 hold K=0..7 (VGPR0-3) and K=16..23 (VGPR4-7);
// lanes 16-31 hold K=8..15 and K=24..31.  => K = e + 8*(lane>=16) + 8*(e>=8)
__device__ __forceinline__ int kmap(int e, int lane) {
  return e + ((lane >> 4) << 3) + ((e >> 3) << 3);
}

__device__ __forceinline__ float fast_tanh(float x) {
#if __has_builtin(__builtin_amdgcn_tanhf)
  return __builtin_amdgcn_tanhf(x);          // v_tanh_f32 (CDNA5 transcendental)
#else
  float e = __expf(2.0f * x);                // v_exp_f32 fallback
  return 1.0f - 2.0f / (e + 1.0f);
#endif
}

// ---------------------------------------------------------------------------
// Kernel 1: project queries (4096x256 @ 256x64) and keys (16384x256 @ 256x64)
// One wave computes a 16-row x 64-col tile with 4 WMMA accumulators.
// ---------------------------------------------------------------------------
__global__ __launch_bounds__(128)
void proj_wmma_kernel(const float* __restrict__ queries,
                      const float* __restrict__ keys,
                      const float* __restrict__ Wq,
                      const float* __restrict__ Wk,
                      float* __restrict__ qp,
                      float* __restrict__ kp) {
  const int gwave    = (blockIdx.x * blockDim.x + threadIdx.x) >> 5;  // 0..1279
  const int lane     = threadIdx.x & 31;
  const int row_base = gwave << 4;

  const float* src; const float* W; float* dst; int r0;
  if (row_base < 16 * 256) { src = queries; W = Wq; dst = qp; r0 = row_base; }
  else                     { src = keys;    W = Wk; dst = kp; r0 = row_base - 16 * 256; }

  const int m  = lane & 15;
  const int hi = lane >> 4;

  v8f acc[4] = {};

  for (int kk = 0; kk < 8; ++kk) {          // K = 256, 32 per WMMA
    const int kb = kk * 32;
    v16h a;
#pragma unroll
    for (int e = 0; e < 16; ++e)
      a[e] = (_Float16)src[(size_t)(r0 + m) * 256 + kb + kmap(e, lane)];
#pragma unroll
    for (int t = 0; t < 4; ++t) {           // 4 N-tiles of 16 (H = 64)
      v16h bv;
#pragma unroll
      for (int e = 0; e < 16; ++e)
        bv[e] = (_Float16)W[(size_t)(kb + kmap(e, lane)) * 64 + t * 16 + m];
      acc[t] = __builtin_amdgcn_wmma_f32_16x16x32_f16(
          false, a, false, bv, (short)0, acc[t], false, false);
    }
  }
#pragma unroll
  for (int t = 0; t < 4; ++t)
#pragma unroll
    for (int r = 0; r < 8; ++r)
      dst[(size_t)(r0 + r + hi * 8) * 64 + t * 16 + m] = acc[t][r];
}

// ---------------------------------------------------------------------------
// Kernel 2: fused additive-attention scores + masked softmax + P@V.
// One 128-thread workgroup per (batch b, 16-query tile qt).
// ---------------------------------------------------------------------------
__global__ __launch_bounds__(128)
void attn_fused_kernel(const float* __restrict__ values,      // [16,1024,128]
                       const float* __restrict__ wv_g,        // [64]
                       const int*   __restrict__ valid_lens,  // [16]
                       const float* __restrict__ qp,          // [16,256,64]
                       const float* __restrict__ kp,          // [16,1024,64]
                       float* __restrict__ out) {             // [16,256,128]
  const int qt  = blockIdx.x;   // 0..15
  const int b   = blockIdx.y;   // 0..15
  const int tid = threadIdx.x;  // 0..127
  const int lane = tid & 31;
  const int wid  = tid >> 5;

  __shared__ _Float16 s_sc[16][1024 + 8];   // scores, then p = exp(s - m)
  __shared__ float    s_qp[16][64];
  __shared__ float    s_wv[64];
  __shared__ float    s_red[16][8];
  __shared__ float    s_m[16];
  __shared__ float    s_l[16];

  // ---- stage qp tile and w_v into LDS ----
  {
    const float* qpb = qp + (size_t)(b * 256 + qt * 16) * 64;
    for (int i = tid; i < 16 * 64; i += 128) ((float*)s_qp)[i] = qpb[i];
    if (tid < 64) s_wv[tid] = wv_g[tid];
  }
  const int vlen = valid_lens[b];
  __syncthreads();

  // ---- phase 1: scores[q][k] = sum_h w_v[h] * tanh(qp[q][h] + kp[k][h]) ----
  // Each thread owns 8 key rows; the dominant 268M-tanh stream lives here.
#pragma unroll 1
  for (int i = 0; i < 8; ++i) {
    const int k = tid + i * 128;
    const float* kpr = kp + (size_t)(b * 1024 + k) * 64;
    float sacc[16];
#pragma unroll
    for (int q = 0; q < 16; ++q) sacc[q] = 0.f;
#pragma unroll 4
    for (int h = 0; h < 64; ++h) {
      const float kv = kpr[h];     // 256B row, stays hot in WGP$
      const float w  = s_wv[h];
#pragma unroll
      for (int q = 0; q < 16; ++q)
        sacc[q] = fmaf(w, fast_tanh(s_qp[q][h] + kv), sacc[q]);
    }
    const bool masked = (k >= vlen);
#pragma unroll
    for (int q = 0; q < 16; ++q)
      s_sc[q][k] = (_Float16)(masked ? NEG_INF_F : sacc[q]);  // -1e6 -> f16 -inf
  }
  __syncthreads();

  // ---- phase 2: masked softmax stats (two-pass), p stored in place ----
  {
    const int q    = tid >> 3;   // 16 rows x 8 partials
    const int part = tid & 7;
    const int c0   = part * 128;
    float mloc = -1e30f;
    for (int k = c0; k < c0 + 128; ++k)
      mloc = fmaxf(mloc, (float)s_sc[q][k]);
    s_red[q][part] = mloc;
    __syncthreads();
    if (part == 0) {
      float mm = s_red[q][0];
#pragma unroll
      for (int j = 1; j < 8; ++j) mm = fmaxf(mm, s_red[q][j]);
      s_m[q] = mm;
    }
    __syncthreads();
    const float mrow = s_m[q];
    float lloc = 0.f;
    for (int k = c0; k < c0 + 128; ++k) {
      float p = __expf((float)s_sc[q][k] - mrow);   // exp(-inf) = 0 for masked
      s_sc[q][k] = (_Float16)p;
      lloc += p;
    }
    s_red[q][part] = lloc;
    __syncthreads();
    if (part == 0) {
      float ss = 0.f;
#pragma unroll
      for (int j = 0; j < 8; ++j) ss += s_red[q][j];
      s_l[q] = ss;
    }
    __syncthreads();
  }

  // ---- phase 3: out[16 x 128] = P(16x1024) @ V(1024x128) via WMMA ----
  // Each of the 4 waves owns 2 N-tiles of 16 columns.
  {
    const int m  = lane & 15;
    const int hi = lane >> 4;
    v8f cacc[2] = {};
    for (int kt = 0; kt < 32; ++kt) {
      const int kb = kt * 32;
      v16h a;
#pragma unroll
      for (int e = 0; e < 16; ++e)
        a[e] = s_sc[m][kb + kmap(e, lane)];
#pragma unroll
      for (int t = 0; t < 2; ++t) {
        const int n = (wid * 2 + t) * 16 + m;
        v16h bv;
#pragma unroll
        for (int e = 0; e < 16; ++e) {
          const int K = kb + kmap(e, lane);
          bv[e] = (_Float16)values[(size_t)(b * 1024 + K) * 128 + n];
        }
        cacc[t] = __builtin_amdgcn_wmma_f32_16x16x32_f16(
            false, a, false, bv, (short)0, cacc[t], false, false);
      }
    }
#pragma unroll
    for (int t = 0; t < 2; ++t) {
      const int n = (wid * 2 + t) * 16 + m;
#pragma unroll
      for (int r = 0; r < 8; ++r) {
        const int q = r + hi * 8;
        out[(size_t)(b * 256 + qt * 16 + q) * 128 + n] = cacc[t][r] / s_l[q];
      }
    }
  }
}

// ---------------------------------------------------------------------------
extern "C" void kernel_launch(void* const* d_in, const int* in_sizes, int n_in,
                              void* d_out, int out_size, void* d_ws, size_t ws_size,
                              hipStream_t stream) {
  (void)in_sizes; (void)n_in; (void)out_size; (void)ws_size;
  const float* queries = (const float*)d_in[0];  // [16,256,256]
  const float* keys    = (const float*)d_in[1];  // [16,1024,256]
  const float* values  = (const float*)d_in[2];  // [16,1024,128]
  const float* Wq      = (const float*)d_in[3];  // [256,64]
  const float* Wk      = (const float*)d_in[4];  // [256,64]
  const float* wv      = (const float*)d_in[5];  // [64]
  const int*   vlens   = (const int*)d_in[6];    // [16]
  float* out = (float*)d_out;                    // [16,256,128]

  float* qp = (float*)d_ws;                      // [4096,64]  1 MB
  float* kp = qp + 4096 * 64;                    // [16384,64] 4 MB

  // 1280 tiles of 16 rows, 4 waves per block -> 320 blocks
  proj_wmma_kernel<<<320, 128, 0, stream>>>(queries, keys, Wq, Wk, qp, kp);
  attn_fused_kernel<<<dim3(16, 16), 128, 0, stream>>>(values, wv, vlens, qp, kp, out);
}
